// GCN_19207093748634
// MI455X (gfx1250) — matrix operand order repaired and checked by reading, not compiled, not run on hardware
//
#include <hip/hip_runtime.h>

// Problem constants from the reference
#define NND   30000
#define NEDGE 960000
#define KHOP  3
#define INF0  94
#define H1D   512
#define H2D   256
#define CD    32

typedef __attribute__((ext_vector_type(16))) _Float16 v16h;
typedef __attribute__((ext_vector_type(8)))  _Float16 v8h;
typedef __attribute__((ext_vector_type(8)))  float    v8f;

// ---------------------------------------------------------------- utilities
__global__ void fzero_k(float* __restrict__ p, long n) {
    long i = (long)blockIdx.x * blockDim.x + threadIdx.x;
    if (i < n) p[i] = 0.0f;
}

__global__ void degree_k(const long long* __restrict__ col, float* __restrict__ deg, int e) {
    int i = blockIdx.x * blockDim.x + threadIdx.x;
    if (i < e) atomicAdd(&deg[(int)col[i]], 1.0f);
}

__global__ void dinv_k(float* __restrict__ deg, int n) {
    int i = blockIdx.x * blockDim.x + threadIdx.x;
    if (i < n) { float d = deg[i]; deg[i] = (d > 0.0f) ? rsqrtf(d) : 0.0f; }
}

__global__ void norm_k(const long long* __restrict__ row, const long long* __restrict__ col,
                       const float* __restrict__ dinv, float* __restrict__ nrm, int e) {
    int i = blockIdx.x * blockDim.x + threadIdx.x;
    if (i < e) nrm[i] = dinv[(int)row[i]] * dinv[(int)col[i]];
}

// h -> f16 slab inside the fused-K A matrix: A16[row][slabOff + k]
__global__ void cvt_a_slab_k(const float* __restrict__ h, _Float16* __restrict__ a16,
                             int kpad, int slab_off, int ktot, long total) {
    long i = (long)blockIdx.x * blockDim.x + threadIdx.x;
    if (i >= total) return;
    long r = i / kpad; int k = (int)(i % kpad);
    a16[r * (long)ktot + slab_off + k] = (_Float16)h[i];
}

// stacked transposed weights: BT[n][m*kpad + k] = w[m][k][n], zero-padded k>=K
__global__ void cvt_w_k(const float* __restrict__ w, _Float16* __restrict__ bt,
                        int K, int nl, int kpad, long total) {
    long i = (long)blockIdx.x * blockDim.x + threadIdx.x;
    if (i >= total) return;
    int ktot = (KHOP + 1) * kpad;
    long n = i / ktot; int rem = (int)(i % ktot);
    int m = rem / kpad, k = rem % kpad;
    bt[i] = (k < K) ? (_Float16)w[((long)m * K + k) * nl + n] : (_Float16)0.0f;
}

// ------------------------------------------------------------ CNN encoder
// one block of 256 threads per node; all intermediates in LDS
__global__ __launch_bounds__(256) void cnn_encoder_k(
    const float* __restrict__ in,
    const float* __restrict__ W1, const float* __restrict__ b1,
    const float* __restrict__ W2, const float* __restrict__ b2,
    const float* __restrict__ W3, const float* __restrict__ b3,
    float* __restrict__ h0 /* [N][96], cols 94,95 zero */) {
    __shared__ float s_in[7 * 98];        // 686
    __shared__ float s_c1[10 * 5 * 96];   // 4800
    __shared__ float s_c2[20 * 3 * 95];   // 5700
    __shared__ float s_w1[90], s_b1[10], s_w2[1200], s_b2[20], s_w3[120], s_b3[1];

    const int tid = threadIdx.x;
    const float* src = in + (long)blockIdx.x * 686;
    for (int i = tid; i < 686;  i += 256) s_in[i] = src[i];
    for (int i = tid; i < 90;   i += 256) s_w1[i] = W1[i];
    for (int i = tid; i < 10;   i += 256) s_b1[i] = b1[i];
    for (int i = tid; i < 1200; i += 256) s_w2[i] = W2[i];
    for (int i = tid; i < 20;   i += 256) s_b2[i] = b2[i];
    for (int i = tid; i < 120;  i += 256) s_w3[i] = W3[i];
    if (tid == 0) s_b3[0] = b3[0];
    __syncthreads();

    // conv1 3x3 VALID -> [10][5][96], leaky_relu(0.01)
    for (int i = tid; i < 4800; i += 256) {
        int x = i % 96, y = (i / 96) % 5, o = i / 480;
        float acc = s_b1[o];
        #pragma unroll
        for (int ky = 0; ky < 3; ky++)
            #pragma unroll
            for (int kx = 0; kx < 3; kx++)
                acc += s_in[(y + ky) * 98 + x + kx] * s_w1[o * 9 + ky * 3 + kx];
        s_c1[i] = (acc > 0.0f) ? acc : 0.01f * acc;
    }
    __syncthreads();

    // conv2 3x2 VALID -> [20][3][95], leaky_relu
    for (int i = tid; i < 5700; i += 256) {
        int x = i % 95, y = (i / 95) % 3, o = i / 285;
        float acc = s_b2[o];
        for (int ci = 0; ci < 10; ci++)
            #pragma unroll
            for (int ky = 0; ky < 3; ky++)
                #pragma unroll
                for (int kx = 0; kx < 2; kx++)
                    acc += s_c1[ci * 480 + (y + ky) * 96 + x + kx]
                         * s_w2[((o * 10 + ci) * 3 + ky) * 2 + kx];
        s_c2[i] = (acc > 0.0f) ? acc : 0.01f * acc;
    }
    __syncthreads();

    // conv3 3x2 VALID -> [94], tanh, write padded to 96
    for (int x = tid; x < 96; x += 256) {
        float v = 0.0f;
        if (x < 94) {
            float acc = s_b3[0];
            for (int ci = 0; ci < 20; ci++)
                #pragma unroll
                for (int ky = 0; ky < 3; ky++)
                    #pragma unroll
                    for (int kx = 0; kx < 2; kx++)
                        acc += s_c2[ci * 285 + ky * 95 + x + kx] * s_w3[(ci * 3 + ky) * 2 + kx];
            v = tanhf(acc);
        }
        h0[(long)blockIdx.x * 96 + x] = v;
    }
}

// ---------------------------------------------------- sparse propagation
// out[col] += norm * h[row]; one block per edge, threads over features
__global__ void propagate_k(const float* __restrict__ h, float* __restrict__ out,
                            const long long* __restrict__ row, const long long* __restrict__ col,
                            const float* __restrict__ nrm, int F) {
    int e = blockIdx.x;
    int r = (int)row[e], c = (int)col[e];
    float w = nrm[e];
    const float* hr = h + (long)r * F;
    float* oc = out + (long)c * F;
    for (int f = threadIdx.x; f < F; f += blockDim.x)
        atomicAdd(&oc[f], w * hr[f]);
}

// ---------------------------------------------------------- WMMA GEMM
// C[M][Nc] = A16[M][Ktot] * BT16[Nc][Ktot]^T   (f16 in, f32 acc)
// block = 4 waves; each wave: one 16-row tile x NT 16-col tiles
template <int NT>
__global__ __launch_bounds__(128) void wmma_gemm_k(
    const _Float16* __restrict__ A, const _Float16* __restrict__ BT,
    float* __restrict__ C, int M, int Nc, int Ktot) {
    const int lane = threadIdx.x & 31;
    const int wid  = threadIdx.x >> 5;
    const int mtile = blockIdx.x * 4 + wid;
    if (mtile * 16 >= M) return;
    const int nbase = blockIdx.y * (16 * NT);

    const int half  = lane >> 4;
    const int mrow  = mtile * 16 + (lane & 15);
    const int aoff  = half * 16;        // A: contiguous K [16h, 16h+16)
    const int ncol  = lane & 15;
    const int boff  = half * 16;        // B: contiguous K [16h, 16h+16)

    v8f acc[NT];
    #pragma unroll
    for (int t = 0; t < NT; t++)
        #pragma unroll
        for (int i = 0; i < 8; i++) acc[t][i] = 0.0f;

    const _Float16* arow = A + (long)mrow * Ktot;
    for (int k0 = 0; k0 < Ktot; k0 += 32) {
        v8h alo = *(const v8h*)(arow + k0 + aoff);
        v8h ahi = *(const v8h*)(arow + k0 + 8 + aoff);
        v16h a = __builtin_shufflevector(alo, ahi, 0, 1, 2, 3, 4, 5, 6, 7,
                                                   8, 9, 10, 11, 12, 13, 14, 15);
        #pragma unroll
        for (int t = 0; t < NT; t++) {
            const _Float16* bp = BT + (long)(nbase + t * 16 + ncol) * Ktot + k0 + boff;
            v16h b = *(const v16h*)bp;
            acc[t] = __builtin_amdgcn_wmma_f32_16x16x32_f16(
                false, a, false, b, (short)0, acc[t], false, false);
        }
    }

    const int r0 = mtile * 16 + 8 * half;   // C: VGPR i -> row r0+i, col = lane&15
    #pragma unroll
    for (int t = 0; t < NT; t++) {
        int cn = nbase + t * 16 + ncol;
        #pragma unroll
        for (int i = 0; i < 8; i++)
            C[(long)(r0 + i) * Nc + cn] = acc[t][i];
    }
}

// ------------------------------------------------------- epilogues
__global__ void bias_tanh_k(const float* __restrict__ in, const float* __restrict__ b,
                            float* __restrict__ out, int nc, long total) {
    long i = (long)blockIdx.x * blockDim.x + threadIdx.x;
    if (i < total) out[i] = tanhf(in[i] + b[(int)(i % nc)]);
}

__global__ void softmax32_k(const float* __restrict__ in, const float* __restrict__ b,
                            float* __restrict__ out, int n) {
    int i = blockIdx.x * blockDim.x + threadIdx.x;
    if (i >= n) return;
    float v[32], mx = -3.4e38f;
    #pragma unroll
    for (int j = 0; j < 32; j++) { v[j] = in[i * 32 + j] + b[j]; mx = fmaxf(mx, v[j]); }
    float s = 0.0f;
    #pragma unroll
    for (int j = 0; j < 32; j++) { v[j] = expf(v[j] - mx); s += v[j]; }
    float inv = 1.0f / s;
    #pragma unroll
    for (int j = 0; j < 32; j++) out[i * 32 + j] = v[j] * inv;
}

// ------------------------------------------------------------------ driver
extern "C" void kernel_launch(void* const* d_in, const int* in_sizes, int n_in,
                              void* d_out, int out_size, void* d_ws, size_t ws_size,
                              hipStream_t stream) {
    const float*     inputs = (const float*)d_in[0];
    const long long* ei     = (const long long*)d_in[1];
    const long long* row    = ei;
    const long long* col    = ei + NEDGE;
    const float* W1 = (const float*)d_in[2],  *b1 = (const float*)d_in[3];
    const float* W2 = (const float*)d_in[4],  *b2 = (const float*)d_in[5];
    const float* W3 = (const float*)d_in[6],  *b3 = (const float*)d_in[7];
    const float* tw[3] = { (const float*)d_in[8],  (const float*)d_in[10], (const float*)d_in[12] };
    const float* tb[3] = { (const float*)d_in[9],  (const float*)d_in[11], (const float*)d_in[13] };

    // workspace carve-up (all offsets 256B aligned)
    char* ws = (char*)d_ws;
    size_t off = 0;
    float* bufA = (float*)(ws + off);  off += (size_t)NND * 512 * sizeof(float);   // 61.44 MB
    float* bufB = (float*)(ws + off);  off += (size_t)NND * 512 * sizeof(float);   // 61.44 MB
    _Float16* a16 = (_Float16*)(ws + off); off += (size_t)NND * 4 * 512 * sizeof(_Float16); // 122.9 MB
    _Float16* w16 = (_Float16*)(ws + off); off += (size_t)512 * 4 * 512 * sizeof(_Float16); // 2 MB
    float* nrm = (float*)(ws + off);   off += (size_t)NEDGE * sizeof(float);
    float* deg = (float*)(ws + off);   off += (size_t)NND * sizeof(float);

    // --- graph normalization: deg -> dinv -> norm ---
    fzero_k<<<(NND + 255) / 256, 256, 0, stream>>>(deg, NND);
    degree_k<<<(NEDGE + 255) / 256, 256, 0, stream>>>(col, deg, NEDGE);
    dinv_k<<<(NND + 255) / 256, 256, 0, stream>>>(deg, NND);
    norm_k<<<(NEDGE + 255) / 256, 256, 0, stream>>>(row, col, deg, nrm, NEDGE);

    // --- CNN encoder -> bufA [N][96] (padded) ---
    cnn_encoder_k<<<NND, 256, 0, stream>>>(inputs, W1, b1, W2, b2, W3, b3, bufA);

    float* hc = bufA;   // current hop buffer
    float* ht = bufB;

    const int Kdims[3]    = { INF0, H1D, H2D };
    const int Kpads[3]    = { 96,   H1D, H2D };
    const int Ndims[3]    = { H1D,  H2D, CD  };

    for (int L = 0; L < 3; L++) {
        const int K = Kdims[L], Kpad = Kpads[L], Nl = Ndims[L];
        const int Ktot = (KHOP + 1) * Kpad;
        const long atot = (long)NND * Kpad;
        const long wtot = (long)Nl * Ktot;

        // stacked transposed f16 weights
        cvt_w_k<<<(int)((wtot + 255) / 256), 256, 0, stream>>>(tw[L], w16, K, Nl, Kpad, wtot);
        // slab 0 = h itself
        cvt_a_slab_k<<<(int)((atot + 255) / 256), 256, 0, stream>>>(hc, a16, Kpad, 0, Ktot, atot);

        // hops 1..3: h <- Â h, stash each as an f16 slab
        const int bdim = (Kpad >= 256) ? 256 : Kpad;   // 96 threads (3 waves) for layer 1
        for (int hop = 1; hop <= KHOP; hop++) {
            fzero_k<<<(int)((atot + 255) / 256), 256, 0, stream>>>(ht, atot);
            propagate_k<<<NEDGE, bdim, 0, stream>>>(hc, ht, row, col, nrm, Kpad);
            float* t = hc; hc = ht; ht = t;
            cvt_a_slab_k<<<(int)((atot + 255) / 256), 256, 0, stream>>>(hc, a16, Kpad, hop * Kpad, Ktot, atot);
        }

        // single fused GEMM over Ktot = 4*Kpad : C = [h0|h1|h2|h3] * [W0;W1;W2;W3]
        const int mtiles = (NND + 15) / 16;          // 1875 exactly
        dim3 blk(128, 1, 1);
        if (Nl >= 64) {
            dim3 grd((mtiles + 3) / 4, Nl / 64, 1);
            wmma_gemm_k<4><<<grd, blk, 0, stream>>>(a16, w16, ht, NND, Nl, Ktot);
        } else {
            dim3 grd((mtiles + 3) / 4, Nl / 32, 1);
            wmma_gemm_k<2><<<grd, blk, 0, stream>>>(a16, w16, ht, NND, Nl, Ktot);
        }

        if (L < 2) {
            const long tot = (long)NND * Nl;
            bias_tanh_k<<<(int)((tot + 255) / 256), 256, 0, stream>>>(ht, tb[L], hc, Nl, tot);
            // hc now holds next layer's input (stride Nl == next Kpad); ht is scratch
        } else {
            softmax32_k<<<(NND + 255) / 256, 256, 0, stream>>>(ht, tb[L], (float*)d_out, NND);
        }
    }
}
